// LinearPps_76381698392174
// MI455X (gfx1250) — matrix operand order repaired and checked by reading, not compiled, not run on hardware
//
#include <hip/hip_runtime.h>

typedef __attribute__((ext_vector_type(16))) __bf16 v16bf;
typedef __attribute__((ext_vector_type(8)))  float  v8f;

#define IN_F       1024
#define OUT_F      1024
#define NUM_TASKS  8
#define NUM_CAT    16
#define VAR_SIZE   256
#define NUM_VARS   4096
#define BATCH      1024

// workspace layout (bytes)
#define OFF_ONEHOT  ((size_t)0)                    // 8*4096*16 f32 = 2 MB
#define OFF_SBIAS   ((size_t)2*1024*1024)          // 8*1024 f32 = 32 KB
#define OFF_AHI     ((size_t)4*1024*1024)          // 16 MB of bf16 frags
#define OFF_ALO     ((size_t)20*1024*1024)
#define OFF_BHI     ((size_t)36*1024*1024)
#define OFF_BLO     ((size_t)52*1024*1024)         // ends at 68 MB

// ---------------------------------------------------------------------------
// 1. Gumbel-softmax: onehot[t][v][c]
__global__ void k_gumbel(const float* __restrict__ uniforms,
                         const float* __restrict__ dist_probs,
                         const float* __restrict__ temp_p,
                         float* __restrict__ onehot) {
    int row = blockIdx.x * blockDim.x + threadIdx.x;       // t*4096+v
    if (row >= NUM_TASKS * NUM_VARS) return;
    const float eps = 1.1920929e-07f;
    float invT = 1.0f / temp_p[0];
    float s[NUM_CAT];
    float mx = -1e30f;
    #pragma unroll
    for (int c = 0; c < NUM_CAT; ++c) {
        float u = uniforms[row * NUM_CAT + c];
        u = fminf(fmaxf(u, eps), 1.0f - eps);
        float g = -__logf(-__logf(u));
        float sc = (dist_probs[row * NUM_CAT + c] + g) * invT;
        s[c] = sc;
        mx = fmaxf(mx, sc);
    }
    float sum = 0.0f;
    #pragma unroll
    for (int c = 0; c < NUM_CAT; ++c) { s[c] = __expf(s[c] - mx); sum += s[c]; }
    float inv = 1.0f / sum;
    #pragma unroll
    for (int c = 0; c < NUM_CAT; ++c) onehot[row * NUM_CAT + c] = s[c] * inv;
}

// ---------------------------------------------------------------------------
// 2. zero sbias + loss
__global__ void k_zero(float* __restrict__ sbias, float* __restrict__ loss) {
    int i = blockIdx.x * blockDim.x + threadIdx.x;
    if (i < NUM_TASKS * OUT_F) sbias[i] = 0.0f;
    if (i == 0) loss[0] = 0.0f;
}

// ---------------------------------------------------------------------------
// 3. Pack permuted/mixed weight into WMMA B-fragment layout (bf16 hi/lo).
//    B frag (32K x 16N tile): lane = (kk>>4)*16 + n, elem j = kk&15.
//    frag index = (t*64 + nt)*32 + kt ; frag = 32 lanes * 16 bf16.
__global__ void k_packB(const long long* __restrict__ perm,
                        const float* __restrict__ weight,
                        const float* __restrict__ onehot,
                        __bf16* __restrict__ Bhi, __bf16* __restrict__ Blo) {
    int p = blockIdx.x * blockDim.x + threadIdx.x;         // 0 .. 1M-1
    if (p >= IN_F * OUT_F) return;
    long long idx = perm[p];
    int v = (int)(idx >> 8);
    int sIdx = (int)(idx & 255);
    float wrow[NUM_CAT];
    #pragma unroll
    for (int c = 0; c < NUM_CAT; ++c)
        wrow[c] = weight[((size_t)v * VAR_SIZE + sIdx) * NUM_CAT + c];

    int i  = p >> 10;          // row in IN_F (K dim)
    int o  = p & 1023;         // col in OUT_F (N dim)
    int kt = i >> 5, kk = i & 31;
    int nt = o >> 4, ni = o & 15;
    int lane = ((kk >> 4) << 4) + ni;
    int j    = kk & 15;
    #pragma unroll
    for (int t = 0; t < NUM_TASKS; ++t) {
        float acc = 0.0f;
        #pragma unroll
        for (int c = 0; c < NUM_CAT; ++c)
            acc += onehot[((size_t)t * NUM_VARS + v) * NUM_CAT + c] * wrow[c];
        __bf16 hi = (__bf16)acc;
        __bf16 lo = (__bf16)(acc - (float)hi);
        size_t base = ((((size_t)(t * 64 + nt)) * 32 + kt) * 32 + lane) * 16 + j;
        Bhi[base] = hi;
        Blo[base] = lo;
    }
}

// ---------------------------------------------------------------------------
// 4. Pack x into WMMA A-fragment layout (bf16 hi/lo).
//    A frag (16M x 32K): lane = (kk>>4)*16 + m, elem j = kk&15
__global__ void k_packA(const float* __restrict__ x,
                        __bf16* __restrict__ Ahi, __bf16* __restrict__ Alo) {
    size_t e = (size_t)blockIdx.x * blockDim.x + threadIdx.x;  // 8M elems
    if (e >= (size_t)BATCH * NUM_TASKS * IN_F) return;
    int n = (int)(e >> 13);            // /(8*1024)
    int r = (int)(e & 8191);
    int t = r >> 10;
    int i = r & 1023;
    float val = x[e];
    __bf16 hi = (__bf16)val;
    __bf16 lo = (__bf16)(val - (float)hi);
    int mt = n >> 4, mi = n & 15;
    int kt = i >> 5, kk = i & 31;
    int lane = ((kk >> 4) << 4) + mi;
    int j    = kk & 15;
    size_t base = ((((size_t)(t * 64 + mt)) * 32 + kt) * 32 + lane) * 16 + j;
    Ahi[base] = hi;
    Alo[base] = lo;
}

// ---------------------------------------------------------------------------
// 5. sampled_bias[t][o] = sum_{v,c} onehot[t,v,c] * bias[v,c,o]   (256 MB stream)
//    grid: 64 v-slices x 8 o-chunks ; 128 threads, fully coalesced over o.
__global__ void k_bias(const float* __restrict__ bias,
                       const float* __restrict__ onehot,
                       float* __restrict__ sbias) {
    int vs = blockIdx.x >> 3;          // v slice (64 vars)
    int oc = blockIdx.x & 7;           // o chunk (128 outs)
    int o  = oc * 128 + threadIdx.x;
    float acc[NUM_TASKS];
    #pragma unroll
    for (int t = 0; t < NUM_TASKS; ++t) acc[t] = 0.0f;
    for (int dv = 0; dv < 64; ++dv) {
        int v = vs * 64 + dv;
        #pragma unroll
        for (int c = 0; c < NUM_CAT; ++c) {
            float bv = bias[((size_t)v * NUM_CAT + c) * OUT_F + o];
            #pragma unroll
            for (int t = 0; t < NUM_TASKS; ++t)
                acc[t] += onehot[((size_t)t * NUM_VARS + v) * NUM_CAT + c] * bv;
        }
    }
    #pragma unroll
    for (int t = 0; t < NUM_TASKS; ++t)
        atomicAdd(&sbias[t * OUT_F + o], acc[t]);
}

// ---------------------------------------------------------------------------
// 6. Main batched GEMM: split-bf16 (hi*hi + hi*lo + lo*hi) WMMA.
//    One wave owns a 16(M) x 64(N) C tile: 4 accumulators, A-frag reused 4x.
__global__ void k_gemm(const __bf16* __restrict__ Ahi, const __bf16* __restrict__ Alo,
                       const __bf16* __restrict__ Bhi, const __bf16* __restrict__ Blo,
                       const float* __restrict__ sbias,
                       float* __restrict__ out) {
    int wave = threadIdx.x >> 5;
    int lane = threadIdx.x & 31;
    int gw   = blockIdx.x * (blockDim.x >> 5) + wave;   // 0..8191
    int t    = gw >> 10;                                // 64*16 tiles per task
    int rem  = gw & 1023;
    int mt   = rem >> 4;                                // M tile (16 rows)
    int ng   = rem & 15;                                // group of 4 N tiles

    // fragment pointers in v16bf units: frag = 32 v16bf, lane picks one.
    const v16bf* Ah = (const v16bf*)Ahi + ((size_t)(t * 64 + mt) * 32) * 32 + lane;
    const v16bf* Al = (const v16bf*)Alo + ((size_t)(t * 64 + mt) * 32) * 32 + lane;
    const v16bf* BhBase = (const v16bf*)Bhi + ((size_t)(t * 64 + ng * 4) * 32) * 32 + lane;
    const v16bf* BlBase = (const v16bf*)Blo + ((size_t)(t * 64 + ng * 4) * 32) * 32 + lane;

    v8f acc[4];
    #pragma unroll
    for (int q = 0; q < 4; ++q) acc[q] = (v8f){0,0,0,0,0,0,0,0};

    for (int kt = 0; kt < 32; ++kt) {
        v16bf ah = Ah[kt * 32];
        v16bf al = Al[kt * 32];
        #pragma unroll
        for (int q = 0; q < 4; ++q) {
            v16bf bh = BhBase[(size_t)q * 1024 + kt * 32];   // nt stride = 32 frags
            v16bf bl = BlBase[(size_t)q * 1024 + kt * 32];
            acc[q] = __builtin_amdgcn_wmma_f32_16x16x32_bf16(
                         false, ah, false, bh, (short)0, acc[q], false, false);
            acc[q] = __builtin_amdgcn_wmma_f32_16x16x32_bf16(
                         false, ah, false, bl, (short)0, acc[q], false, false);
            acc[q] = __builtin_amdgcn_wmma_f32_16x16x32_bf16(
                         false, al, false, bh, (short)0, acc[q], false, false);
        }
    }

    // C/D layout: lane L -> col = n0 + (L&15); VGPR r -> row = r + 8*(L>>4)
    int half = (lane >> 4) << 3;
    #pragma unroll
    for (int q = 0; q < 4; ++q) {
        int col = (ng * 4 + q) * 16 + (lane & 15);
        float b = sbias[t * OUT_F + col];
        #pragma unroll
        for (int r = 0; r < 8; ++r) {
            int row = mt * 16 + half + r;
            out[((size_t)row * NUM_TASKS + t) * OUT_F + col] = acc[q][r] + b;
        }
    }
}

// ---------------------------------------------------------------------------
// 7. Decorrelation loss: per var, off-diagonal Gram of 256x16 codebook.
__global__ void k_loss(const float* __restrict__ weight, float* __restrict__ loss) {
    __shared__ float lw[VAR_SIZE * NUM_CAT];   // 16 KB
    int v = blockIdx.x;
    for (int idx = threadIdx.x; idx < VAR_SIZE * NUM_CAT; idx += blockDim.x)
        lw[idx] = weight[(size_t)v * VAR_SIZE * NUM_CAT + idx];
    __syncthreads();
    if (threadIdx.x < 120) {
        int k = threadIdx.x, c = 0, remn = NUM_CAT - 1;
        while (k >= remn) { k -= remn; ++c; --remn; }
        int d = c + 1 + k;
        float g = 0.0f;
        for (int s = 0; s < VAR_SIZE; ++s)
            g += lw[s * NUM_CAT + c] * lw[s * NUM_CAT + d];
        atomicAdd(loss, 2.0f * g * g);     // symmetric off-diagonal pair
    }
}

// ---------------------------------------------------------------------------
extern "C" void kernel_launch(void* const* d_in, const int* in_sizes, int n_in,
                              void* d_out, int out_size, void* d_ws, size_t ws_size,
                              hipStream_t stream) {
    const float*     x       = (const float*)d_in[0];
    const float*     unifs   = (const float*)d_in[1];
    const float*     temp    = (const float*)d_in[2];
    const float*     dprobs  = (const float*)d_in[3];
    const float*     weight  = (const float*)d_in[4];
    const float*     bias    = (const float*)d_in[5];
    const long long* perm    = (const long long*)d_in[6];

    float* out  = (float*)d_out;                       // [1024,8,1024]
    float* loss = out + (size_t)BATCH * NUM_TASKS * OUT_F;

    char*   ws     = (char*)d_ws;
    float*  onehot = (float*)(ws + OFF_ONEHOT);
    float*  sbias  = (float*)(ws + OFF_SBIAS);
    __bf16* Ahi    = (__bf16*)(ws + OFF_AHI);
    __bf16* Alo    = (__bf16*)(ws + OFF_ALO);
    __bf16* Bhi    = (__bf16*)(ws + OFF_BHI);
    __bf16* Blo    = (__bf16*)(ws + OFF_BLO);

    k_gumbel<<<128, 256, 0, stream>>>(unifs, dprobs, temp, onehot);
    k_zero  <<<32, 256, 0, stream>>>(sbias, loss);
    k_packB <<<4096, 256, 0, stream>>>(perm, weight, onehot, Bhi, Blo);
    k_packA <<<32768, 256, 0, stream>>>(x, Ahi, Alo);
    k_bias  <<<512, 128, 0, stream>>>(bias, onehot, sbias);
    k_gemm  <<<1024, 256, 0, stream>>>(Ahi, Alo, Bhi, Blo, sbias, out);
    k_loss  <<<NUM_VARS, 128, 0, stream>>>(weight, loss);
}
